// RS2G_4733053960344
// MI455X (gfx1250) — matrix operand order repaired and checked by reading, not compiled, not run on hardware
//
#include <hip/hip_runtime.h>
#include <hip/hip_bf16.h>
#include <math.h>
#include <stdint.h>

#define T_  32
#define N_  256
#define FIN 15
#define EH  30
#define NF  64
#define NR  8
#define HID 256
#define D1  256
#define D2  128
#define NC  8
#define LDA 272   // padded LDS row stride (halfs): 544B row = 8-bank step, no serialization

typedef _Float16 h16;
typedef __attribute__((ext_vector_type(16))) _Float16 v16h;
typedef __attribute__((ext_vector_type(8)))  _Float16 v8h;
typedef __attribute__((ext_vector_type(8)))  float    v8f;
typedef int v4i_ __attribute__((vector_size(16)));          // b128 payload type

// ---------------- async global->LDS staging (CDNA5 ASYNCcnt path, guarded) ----------------
#if __has_builtin(__builtin_amdgcn_global_load_async_to_lds_b128) && \
    __has_builtin(__builtin_amdgcn_s_wait_asynccnt)
#define HAVE_ASYNC_LDS 1
#else
#define HAVE_ASYNC_LDS 0
#endif

// copy 16 bytes (8 halfs) from global to LDS, per lane
__device__ __forceinline__ void stage16(const h16* g, h16* l) {
#if HAVE_ASYNC_LDS
  __builtin_amdgcn_global_load_async_to_lds_b128(
      (__attribute__((address_space(1))) v4i_*)(uintptr_t)g,
      (__attribute__((address_space(3))) v4i_*)(uintptr_t)l, 0, 0);
#else
  *(v8h*)l = *(const v8h*)g;
#endif
}
// wait for own async copies, then make LDS visible workgroup-wide
__device__ __forceinline__ void stage_join() {
#if HAVE_ASYNC_LDS
  __builtin_amdgcn_s_wait_asynccnt(0);
#endif
  __syncthreads();
}

// ---------------- WMMA fragment helpers (wave32, 16x16x32 f16) ----------------
// A (16 rows x 32 k): lane holds row = m0 + (lane&15); lanes 0-15 carry K in
// {k..k+7, k+16..k+23}, lanes 16-31 carry {k+8..k+15, k+24..k+31} (ISA 7.12.2).
__device__ __forceinline__ v16h frag_a16(const h16* base, int ld, int m0, int k, int lane) {
  int row = m0 + (lane & 15);
  int kk  = k + ((lane >> 4) << 3);
  const h16* p = base + (size_t)row * ld + kk;
  v8h lo = *(const v8h*)p;
  v8h hi = *(const v8h*)(p + 16);
  return __builtin_shufflevector(lo, hi, 0,1,2,3,4,5,6,7,8,9,10,11,12,13,14,15);
}
// B (32 k x 16 cols), supplied TRANSPOSED (BT[col][k] row-major): lane holds
// column = n0 + (lane&15); lanes 0-15 carry K {k..k+7, k+16..k+23}, lanes 16-31 {k+8..k+15, k+24..k+31}.
__device__ __forceinline__ v16h frag_b16(const h16* baseT, int ld, int n0, int k, int lane) {
  int col = n0 + (lane & 15);
  int kk  = k + ((lane >> 4) << 3);
  const h16* p = baseT + (size_t)col * ld + kk;
  v8h lo = *(const v8h*)p;
  v8h hi = *(const v8h*)(p + 16);
  return __builtin_shufflevector(lo, hi, 0,1,2,3,4,5,6,7,8,9,10,11,12,13,14,15);
}

// ---------------- node encoder + edge projections ----------------
__global__ void k_node_enc(const float* __restrict__ seq,
                           const float* __restrict__ w1, const float* __restrict__ b1,
                           const float* __restrict__ w2, const float* __restrict__ b2,
                           const float* __restrict__ eew1,
                           h16* __restrict__ X0, float* __restrict__ pi, float* __restrict__ pj) {
  int id = blockIdx.x * blockDim.x + threadIdx.x;   // t*N + n
  if (id >= T_ * N_) return;
  float x[FIN];
  #pragma unroll
  for (int f = 0; f < FIN; ++f) x[f] = seq[id * FIN + f];
  float h1[EH];
  for (int c = 0; c < EH; ++c) {
    float s = b1[c];
    #pragma unroll
    for (int f = 0; f < FIN; ++f) s += x[f] * w1[f * EH + c];
    h1[c] = s > 0.f ? s : 0.f;
  }
  for (int c = 0; c < NF; ++c) {
    float s = b2[c];
    #pragma unroll
    for (int k = 0; k < EH; ++k) s += h1[k] * w2[k * NF + c];
    X0[(size_t)id * NF + c] = (h16)(s > 0.f ? s : 0.f);
  }
  for (int c = 0; c < EH; ++c) {
    float si = 0.f, sj = 0.f;
    #pragma unroll
    for (int f = 0; f < FIN; ++f) {
      si += x[f] * eew1[f * EH + c];
      sj += x[f] * eew1[(FIN + f) * EH + c];
    }
    pi[(size_t)id * EH + c] = si;
    pj[(size_t)id * EH + c] = sj;
  }
}

// ---------------- edge encoder -> adjacency A[t][r][i][j] (f16, {0,1,2}) ----------------
__global__ void k_edges(const float* __restrict__ pi, const float* __restrict__ pj,
                        const float* __restrict__ eb1,
                        const float* __restrict__ w2, const float* __restrict__ b2,
                        h16* __restrict__ A) {
  int ti = blockIdx.x;              // t*N + i
  int t = ti >> 8, i = ti & 255;
  int j = threadIdx.x;              // 0..255
  if (j < i) return;
  if (j == i) {
    #pragma unroll
    for (int r = 0; r < NR; ++r)
      A[(((size_t)t * NR + r) * N_ + i) * N_ + i] = (h16)0.f;
    return;
  }
  float eh[EH];
  #pragma unroll
  for (int c = 0; c < EH; ++c) {
    float v = pi[(size_t)ti * EH + c] + pj[((size_t)t * N_ + j) * EH + c] + eb1[c];
    eh[c] = v > 0.f ? v : 0.f;
  }
  float sg[NR];
  #pragma unroll
  for (int r = 0; r < NR; ++r) {
    float s = b2[r];
    #pragma unroll
    for (int c = 0; c < EH; ++c) s += eh[c] * w2[c * NR + r];
    sg[r] = 1.f / (1.f + expf(-s));
  }
  int am = 0; float best = sg[0];
  #pragma unroll
  for (int r = 1; r < NR; ++r) if (sg[r] > best) { best = sg[r]; am = r; }
  #pragma unroll
  for (int r = 0; r < NR; ++r) {
    float v = (r == am ? 1.f : 0.f) + (sg[r] > 0.5f ? 1.f : 0.f);
    h16 hv = (h16)v;
    A[(((size_t)t * NR + r) * N_ + i) * N_ + j] = hv;
    A[(((size_t)t * NR + r) * N_ + j) * N_ + i] = hv;
  }
}

// ---------------- reciprocal relation in-degree: rd[t][r][n] = 1/max(sum_m A,1) ----------------
__global__ void k_deg(const h16* __restrict__ A, float* __restrict__ rd) {
  int id = blockIdx.x * blockDim.x + threadIdx.x;   // (t*NR + r)*N + n
  if (id >= T_ * NR * N_) return;
  const h16* row = A + (size_t)id * N_;
  float s = 0.f;
  for (int m = 0; m < N_; ++m) s += (float)row[m];
  rd[id] = 1.f / fmaxf(s, 1.f);
}

// ---------------- f32 [B][K][H] -> f16 [B][H][K] (weight transpose for B-operands) ----------------
__global__ void k_trans(const float* __restrict__ src, h16* __restrict__ dst,
                        int K, int H, int total) {
  int id = blockIdx.x * blockDim.x + threadIdx.x;
  if (id >= total) return;
  int kh = K * H;
  int b = id / kh, rem = id - b * kh;
  int k = rem / H, h = rem - k * H;
  dst[((size_t)b * H + h) * K + k] = (h16)src[id];
}

// ---------------- WMMA GEMM: xwT[r][t][h][m] = (X[M,K] @ Wr[K,H])^T, f16 out --------------
// Block = 512 threads (16 waves = 16 column tiles) per m-tile; X tile staged in
// LDS once (async DMA) and reused across all 16 column tiles x 8 relations.
__global__ __launch_bounds__(512) void k_gemm_xwT(const h16* __restrict__ X,
                                                  const h16* __restrict__ wT /*[R][H][K]*/,
                                                  h16* __restrict__ xwT, int K) {
  __shared__ __align__(16) h16 Xs[16 * LDA];
  int m0   = blockIdx.x << 4;
  int wave = threadIdx.x >> 5;
  int lane = threadIdx.x & 31;
  int n0   = wave << 4;
  int ldl  = K + 16;
  int nchunk = 2 * K;                       // (16*K)/8 v8h chunks
  for (int c = threadIdx.x; c < nchunk; c += 512) {
    int e = c << 3;
    int srow = e / K, scol = e - srow * K;  // scol multiple of 8
    stage16(X + (size_t)(m0 + srow) * K + scol, Xs + srow * ldl + scol);
  }
  stage_join();
  int col  = n0 + (lane & 15);
  int mrow = m0 + ((lane >> 4) << 3);
  int t = mrow >> 8, ml = mrow & 255;       // 256 rows per frame
  for (int r = 0; r < NR; ++r) {
    const h16* bt = wT + (size_t)r * HID * K;
    __builtin_prefetch(bt + (size_t)col * K, 0, 1);
    v8f acc = {};
    for (int k = 0; k < K; k += 32) {
      v16h a = frag_a16(Xs, ldl, 0, k, lane);
      v16h b = frag_b16(bt, K, n0, k, lane);
      acc = __builtin_amdgcn_wmma_f32_16x16x32_f16(false, a, false, b, (short)0, acc, false, false);
    }
    h16* dst = xwT + (((size_t)r * T_ + t) * HID + col) * N_ + ml;
    v8h o;
    #pragma unroll
    for (int i = 0; i < 8; ++i) o[i] = (h16)acc[i];
    *(v8h*)dst = o;                          // contiguous 16B store
  }
}

// ---------------- WMMA GEMM: out[M][H] = X[M,K] @ W[K,H], f32 out (root term) ----------------
__global__ __launch_bounds__(512) void k_gemm_f32(const h16* __restrict__ X,
                                                  const h16* __restrict__ wT /*[H][K]*/,
                                                  float* __restrict__ out, int K) {
  __shared__ __align__(16) h16 Xs[16 * LDA];
  int m0   = blockIdx.x << 4;
  int wave = threadIdx.x >> 5;
  int lane = threadIdx.x & 31;
  int n0   = wave << 4;
  int ldl  = K + 16;
  int nchunk = 2 * K;
  for (int c = threadIdx.x; c < nchunk; c += 512) {
    int e = c << 3;
    int srow = e / K, scol = e - srow * K;
    stage16(X + (size_t)(m0 + srow) * K + scol, Xs + srow * ldl + scol);
  }
  stage_join();
  int col  = n0 + (lane & 15);
  int row0 = m0 + ((lane >> 4) << 3);
  v8f acc = {};
  for (int k = 0; k < K; k += 32) {
    v16h a = frag_a16(Xs, ldl, 0, k, lane);
    v16h b = frag_b16(wT, K, n0, k, lane);
    acc = __builtin_amdgcn_wmma_f32_16x16x32_f16(false, a, false, b, (short)0, acc, false, false);
  }
  #pragma unroll
  for (int i = 0; i < 8; ++i) out[(size_t)(row0 + i) * HID + col] = acc[i];
}

// ---------------- RGCN message passing: sum_r rd * (A_r @ xw_r) + root + bias, ReLU --------
// Block = 512 threads (16 waves = 16 feature tiles) per (t, node-row tile).
// Adjacency tiles staged through LDS (async DMA) in chunks of 4 relations (34.8KB).
__global__ __launch_bounds__(512) void k_rgcn_msg(const h16* __restrict__ A /*[T][R][N][N]*/,
                                                  const h16* __restrict__ xwT /*[R][T][H][N]*/,
                                                  const float* __restrict__ rd /*[T][R][N]*/,
                                                  const float* __restrict__ rootb /*[T*N][H]*/,
                                                  const float* __restrict__ bias /*[H]*/,
                                                  float* __restrict__ of32,
                                                  h16* __restrict__ of16) {
  __shared__ __align__(16) h16 As[4 * 16 * LDA];   // 34816 bytes
  int t    = blockIdx.x >> 4;
  int m0   = (blockIdx.x & 15) << 4;
  int wave = threadIdx.x >> 5;
  int lane = threadIdx.x & 31;
  int n0   = wave << 4;
  int col  = n0 + (lane & 15);
  int row0 = m0 + ((lane >> 4) << 3);
  v8f tot = {};
  for (int rr = 0; rr < NR; rr += 4) {
    __syncthreads();                                // protect previous chunk
    #pragma unroll
    for (int q = 0; q < 4; ++q) {
      int r = rr + q;
      // 16-row A block is one contiguous 8KB run: rows m0..m0+15 of slice [t][r]
      const h16* src = A + (((size_t)t * NR + r) * N_ + m0) * N_;
      int e = threadIdx.x << 3;                     // 512 threads x 8 halfs = 4096
      int srow = e >> 8, scol = e & 255;
      stage16(src + e, As + (q * 16 + srow) * LDA + scol);
    }
    stage_join();
    #pragma unroll
    for (int q = 0; q < 4; ++q) {
      int r = rr + q;
      const h16* Bs = xwT + ((size_t)r * T_ + t) * (size_t)HID * N_;
      __builtin_prefetch(Bs + (size_t)col * N_, 0, 1);
      const h16* Al = As + q * 16 * LDA;
      v8f acc = {};
      #pragma unroll
      for (int k = 0; k < N_; k += 32) {
        v16h a = frag_a16(Al, LDA, 0, k, lane);
        v16h b = frag_b16(Bs, N_, n0, k, lane);
        acc = __builtin_amdgcn_wmma_f32_16x16x32_f16(false, a, false, b, (short)0, acc, false, false);
      }
      const float* rds = rd + ((size_t)t * NR + r) * N_ + row0;
      #pragma unroll
      for (int i = 0; i < 8; ++i) tot[i] += acc[i] * rds[i];
    }
  }
  float bv = bias[col];
  #pragma unroll
  for (int i = 0; i < 8; ++i) {
    size_t idx = (size_t)(t * N_ + row0 + i) * HID + col;
    float v = tot[i] + rootb[idx] + bv;
    v = v > 0.f ? v : 0.f;
    of32[idx] = v;
    if (of16) of16[idx] = (h16)v;
  }
}

// ---------------- tail: pool / fc1 / temporal mean / fc15 / fc2+logsoftmax ----------------
__global__ void k_pool(const float* __restrict__ x1, const float* __restrict__ x2,
                       float* __restrict__ gp) {
  int id = blockIdx.x * blockDim.x + threadIdx.x;   // t*512 + h
  if (id >= T_ * 2 * HID) return;
  int t = id >> 9, h = id & 511;
  const float* src = (h < HID) ? (x1 + (size_t)t * N_ * HID + h)
                               : (x2 + (size_t)t * N_ * HID + (h - HID));
  float s = 0.f;
  for (int n = 0; n < N_; ++n) s += src[(size_t)n * HID];
  gp[id] = s * (1.f / N_);
}

__global__ void k_fc1(const float* __restrict__ gp, const float* __restrict__ w,
                      const float* __restrict__ b, float* __restrict__ h1) {
  int id = blockIdx.x * blockDim.x + threadIdx.x;   // t*D1 + d
  if (id >= T_ * D1) return;
  int t = id / D1, d = id - t * D1;
  float s = b[d];
  for (int k = 0; k < 2 * HID; ++k) s += gp[t * 2 * HID + k] * w[k * D1 + d];
  h1[id] = s > 0.f ? s : 0.f;
}

__global__ void k_mean(const float* __restrict__ h1, float* __restrict__ hm) {
  int d = threadIdx.x;
  if (d >= D1) return;
  float s = 0.f;
  for (int t = 0; t < T_; ++t) s += h1[t * D1 + d];
  hm[d] = s * (1.f / T_);
}

__global__ void k_fc15(const float* __restrict__ hm, const float* __restrict__ w,
                       const float* __restrict__ b, float* __restrict__ h2) {
  int d = threadIdx.x;
  if (d >= D2) return;
  float s = b[d];
  for (int k = 0; k < D1; ++k) s += hm[k] * w[k * D2 + d];
  h2[d] = s > 0.f ? s : 0.f;
}

__global__ void k_final(const float* __restrict__ h2, const float* __restrict__ w,
                        const float* __restrict__ b, float* __restrict__ out) {
  if (threadIdx.x != 0 || blockIdx.x != 0) return;
  float lg[NC];
  for (int c = 0; c < NC; ++c) {
    float s = b[c];
    for (int k = 0; k < D2; ++k) s += h2[k] * w[k * NC + c];
    lg[c] = s;
  }
  float m = lg[0];
  for (int c = 1; c < NC; ++c) m = fmaxf(m, lg[c]);
  float se = 0.f;
  for (int c = 0; c < NC; ++c) se += expf(lg[c] - m);
  float lse = logf(se);
  for (int c = 0; c < NC; ++c) out[c] = lg[c] - m - lse;
}

// ---------------- host orchestration ----------------
extern "C" void kernel_launch(void* const* d_in, const int* in_sizes, int n_in,
                              void* d_out, int out_size, void* d_ws, size_t ws_size,
                              hipStream_t stream) {
  (void)in_sizes; (void)n_in; (void)out_size; (void)ws_size;
  const float* seq     = (const float*)d_in[0];
  const float* ne_w1   = (const float*)d_in[1];
  const float* ne_b1   = (const float*)d_in[2];
  const float* ne_w2   = (const float*)d_in[3];
  const float* ne_b2   = (const float*)d_in[4];
  const float* ee_w1   = (const float*)d_in[5];
  const float* ee_b1   = (const float*)d_in[6];
  const float* ee_w2   = (const float*)d_in[7];
  const float* ee_b2   = (const float*)d_in[8];
  const float* g1_wrel = (const float*)d_in[9];
  const float* g1_wroot= (const float*)d_in[10];
  const float* g1_b    = (const float*)d_in[11];
  const float* g2_wrel = (const float*)d_in[12];
  const float* g2_wroot= (const float*)d_in[13];
  const float* g2_b    = (const float*)d_in[14];
  const float* fc1_w   = (const float*)d_in[15];
  const float* fc1_b   = (const float*)d_in[16];
  const float* fc15_w  = (const float*)d_in[17];
  const float* fc15_b  = (const float*)d_in[18];
  const float* fc2_w   = (const float*)d_in[19];
  const float* fc2_b   = (const float*)d_in[20];
  float* out = (float*)d_out;

  const int TN = T_ * N_;
  char* ws = (char*)d_ws;
  size_t off = 0;
  auto alloc = [&](size_t bytes) -> char* {
    char* p = ws + off;
    off = (off + bytes + 255) & ~(size_t)255;
    return p;
  };
  float* pi       = (float*)alloc((size_t)TN * EH * 4);
  float* pj       = (float*)alloc((size_t)TN * EH * 4);
  h16*   X0       = (h16*)  alloc((size_t)TN * NF * 2);
  h16*   Abuf     = (h16*)  alloc((size_t)T_ * NR * N_ * N_ * 2);
  float* rdb      = (float*)alloc((size_t)T_ * NR * N_ * 4);
  h16*   g1_wrelT = (h16*)  alloc((size_t)NR * HID * NF * 2);
  h16*   g1_wrootT= (h16*)  alloc((size_t)HID * NF * 2);
  h16*   g2_wrelT = (h16*)  alloc((size_t)NR * HID * HID * 2);
  h16*   g2_wrootT= (h16*)  alloc((size_t)HID * HID * 2);
  h16*   xwT      = (h16*)  alloc((size_t)NR * T_ * HID * N_ * 2);
  float* rootb    = (float*)alloc((size_t)TN * HID * 4);
  float* x1f      = (float*)alloc((size_t)TN * HID * 4);
  h16*   x1h      = (h16*)  alloc((size_t)TN * HID * 2);
  float* x2f      = (float*)alloc((size_t)TN * HID * 4);
  float* gp       = (float*)alloc((size_t)T_ * 2 * HID * 4);
  float* h1b      = (float*)alloc((size_t)T_ * D1 * 4);
  float* hm       = (float*)alloc((size_t)D1 * 4);
  float* h2b      = (float*)alloc((size_t)D2 * 4);

  // encoders + adjacency
  k_node_enc<<<TN / 256, 256, 0, stream>>>(seq, ne_w1, ne_b1, ne_w2, ne_b2, ee_w1, X0, pi, pj);
  k_edges<<<TN, 256, 0, stream>>>(pi, pj, ee_b1, ee_w2, ee_b2, Abuf);
  k_deg<<<(T_ * NR * N_) / 256, 256, 0, stream>>>(Abuf, rdb);

  // weight transposes (f32 -> f16, B-operand layout)
  k_trans<<<(NR * NF * HID + 255) / 256, 256, 0, stream>>>(g1_wrel,  g1_wrelT,  NF,  HID, NR * NF * HID);
  k_trans<<<(NF * HID + 255) / 256, 256, 0, stream>>>(g1_wroot, g1_wrootT, NF,  HID, NF * HID);
  k_trans<<<(NR * HID * HID + 255) / 256, 256, 0, stream>>>(g2_wrel,  g2_wrelT,  HID, HID, NR * HID * HID);
  k_trans<<<(HID * HID + 255) / 256, 256, 0, stream>>>(g2_wroot, g2_wrootT, HID, HID, HID * HID);

  // RGCN layer 1 (K = 64)
  k_gemm_xwT<<<TN / 16, 512, 0, stream>>>(X0, g1_wrelT, xwT, NF);
  k_gemm_f32<<<TN / 16, 512, 0, stream>>>(X0, g1_wrootT, rootb, NF);
  k_rgcn_msg<<<T_ * 16, 512, 0, stream>>>(Abuf, xwT, rdb, rootb, g1_b, x1f, x1h);

  // RGCN layer 2 (K = 256)
  k_gemm_xwT<<<TN / 16, 512, 0, stream>>>(x1h, g2_wrelT, xwT, HID);
  k_gemm_f32<<<TN / 16, 512, 0, stream>>>(x1h, g2_wrootT, rootb, HID);
  k_rgcn_msg<<<T_ * 16, 512, 0, stream>>>(Abuf, xwT, rdb, rootb, g2_b, x2f, (h16*)nullptr);

  // tail
  k_pool<<<(T_ * 2 * HID) / 256, 256, 0, stream>>>(x1f, x2f, gp);
  k_fc1<<<(T_ * D1) / 256, 256, 0, stream>>>(gp, fc1_w, fc1_b, h1b);
  k_mean<<<1, 256, 0, stream>>>(h1b, hm);
  k_fc15<<<1, 128, 0, stream>>>(hm, fc15_w, fc15_b, h2b);
  k_final<<<1, 32, 0, stream>>>(h2b, fc2_w, fc2_b, out);
}
